// MaskedAttention_69578470195511
// MI455X (gfx1250) — matrix operand order repaired and checked by reading, not compiled, not run on hardware
//
#include <hip/hip_runtime.h>

// ---------------------------------------------------------------------------
// MaskedAttention for MI455X (gfx1250), wave32 + WMMA bf16 + async LDS copies.
//
// Shapes: B=4, T=2048, E=H=1024.
// Pipeline:
//   (1) cvt fp32->bf16 : X, Wq, Wk, Wv
//   (2) Q/K/V = Xb @ Wb            (wmma bf16, f32 acc, bf16 store)
//   (3) S = Q @ K^T * E^-0.5       (wmma bf16, f32 store, batched)
//   (4) column softmax over i>=j   (mask==exclude i<j), bf16 P
//   (5) out = P @ V                (wmma bf16, f32 store, batched)
// ---------------------------------------------------------------------------

typedef __attribute__((ext_vector_type(16))) __bf16 bf16x16;
typedef __attribute__((ext_vector_type(8)))  __bf16 bf16x8;
typedef __attribute__((ext_vector_type(4)))  __bf16 bf16x4;
typedef __attribute__((ext_vector_type(8)))  float  f32x8;
typedef __attribute__((ext_vector_type(4)))  int    i32x4;

#define TDIM   2048
#define EDIM   1024
#define HDIM   1024
#define BDIM   4
#define LDSTR  40          // padded LDS row stride in bf16 elements (80B, 16B-aligned)

// ---- gfx1250 async global->LDS copy (direct path, ASYNCcnt), with fallback --
#if defined(__has_builtin)
#if __has_builtin(__builtin_amdgcn_global_load_async_to_lds_b128)
#define ASYNC_LDS 1
#endif
#endif
#ifndef ASYNC_LDS
#define ASYNC_LDS 0
#endif

__device__ __forceinline__ void cp_lds_b128(const __bf16* g, __bf16* l) {
#if ASYNC_LDS
    __builtin_amdgcn_global_load_async_to_lds_b128(
        (__attribute__((address_space(1))) i32x4*)g,
        (__attribute__((address_space(3))) i32x4*)l, 0, 0);
#else
    *(bf16x8*)l = *(const bf16x8*)g;
#endif
}

__device__ __forceinline__ void wait_async_lds() {
#if ASYNC_LDS
#if __has_builtin(__builtin_amdgcn_s_wait_asynccnt)
    __builtin_amdgcn_s_wait_asynccnt(0);
#else
    asm volatile("s_wait_asynccnt 0x0" ::: "memory");
#endif
#endif
}

// ---------------------------------------------------------------------------
// fp32 -> bf16 elementwise conversion (grid-stride)
// ---------------------------------------------------------------------------
__global__ void cvt_f32_bf16(const float* __restrict__ in, __bf16* __restrict__ out, int n) {
    int i = blockIdx.x * blockDim.x + threadIdx.x;
    int stride = gridDim.x * blockDim.x;
    for (; i < n; i += stride) out[i] = (__bf16)in[i];
}

// ---------------------------------------------------------------------------
// WMMA bf16 GEMM: C[M,N] = scale * A[M,K] x B
//   NT=false: B is row-major [K,N]
//   NT=true : B is accessed as Bsrc[n][k] (i.e. C = A * Bsrc^T), Bsrc [N,K]
// Block: 128 threads (4 waves). Block tile 128x128, wave tile 64x64,
// 16 accumulators of 16x16 per wave, K-step 32, double-buffered LDS with
// async global->LDS copies overlapping WMMA compute.
// Batched via blockIdx.z with element strides sA/sB/sC.
// ---------------------------------------------------------------------------
template <bool NT, bool OUT_BF16>
__global__ __launch_bounds__(128)
void gemm_wmma_bf16(const __bf16* __restrict__ Ag, const __bf16* __restrict__ Bg,
                    void* __restrict__ Cg, int Md, int Nd, int Kd,
                    long long sA, long long sB, long long sC, float scale) {
    (void)Md;
    __shared__ __attribute__((aligned(16))) __bf16 As[2][128 * LDSTR];
    __shared__ __attribute__((aligned(16))) __bf16 Bs[2][128 * LDSTR]; // transposed: Bs[n][k]

    const int bz = blockIdx.z;
    const __bf16* A = Ag + (size_t)bz * sA;
    const __bf16* B = Bg + (size_t)bz * sB;

    const int m0 = blockIdx.y * 128;
    const int n0 = blockIdx.x * 128;

    const int tid  = threadIdx.x;
    const int lane = tid & 31;
    const int wave = tid >> 5;          // 0..3
    const int wm   = (wave >> 1) * 64;  // wave tile row origin inside block tile
    const int wn   = (wave & 1) * 64;   // wave tile col origin inside block tile

    const int half = lane >> 4;         // 0 | 1
    const int l16  = lane & 15;
    const int a_koff = half * 8;        // A frag: first run K offset (0 or 8)
    const int b_koff = half * 8;        // B frag: first run K offset (0 or 8)

    f32x8 acc[4][4] = {};

    // stage one 32-deep K slice (A tile + transposed B tile) into buffer `buf`
    auto stage = [&](int k0, int buf) {
        {   // A tile: 128 rows x 32 halves, thread t owns row t (4 async b128)
            const __bf16* src = A + (size_t)(m0 + tid) * Kd + k0;
            __bf16* dst = &As[buf][tid * LDSTR];
            cp_lds_b128(src + 0,  dst + 0);
            cp_lds_b128(src + 8,  dst + 8);
            cp_lds_b128(src + 16, dst + 16);
            cp_lds_b128(src + 24, dst + 24);
        }
        if (NT) {
            // B-column n == row (n0+n) of Bsrc: contiguous in k. Same as A path.
            const __bf16* src = B + (size_t)(n0 + tid) * Kd + k0;
            __bf16* dst = &Bs[buf][tid * LDSTR];
            cp_lds_b128(src + 0,  dst + 0);
            cp_lds_b128(src + 8,  dst + 8);
            cp_lds_b128(src + 16, dst + 16);
            cp_lds_b128(src + 24, dst + 24);
        } else {
            // B row-major [K,N]: 4x4 in-register micro-transpose.
            // chunk c covers k rows [kk,kk+4) x n cols [nn,nn+4).
            #pragma unroll
            for (int p = 0; p < 2; ++p) {
                const int c  = p * 128 + tid;      // 0..255
                const int kk = (c >> 5) * 4;       // 0..28
                const int nn = (c & 31) * 4;       // 0..124
                const __bf16* src = B + (size_t)(k0 + kk) * Nd + n0 + nn;
                bf16x4 r0 = *(const bf16x4*)(src + 0 * Nd);
                bf16x4 r1 = *(const bf16x4*)(src + 1 * Nd);
                bf16x4 r2 = *(const bf16x4*)(src + 2 * Nd);
                bf16x4 r3 = *(const bf16x4*)(src + 3 * Nd);
                #pragma unroll
                for (int e = 0; e < 4; ++e) {
                    bf16x4 t = { r0[e], r1[e], r2[e], r3[e] };
                    *(bf16x4*)&Bs[buf][(nn + e) * LDSTR + kk] = t;   // ds_store_b64
                }
                if (k0 + 64 < Kd)   // prefetch the tile after next (global_prefetch_b8)
                    __builtin_prefetch(src + (size_t)64 * Nd, 0, 0);
            }
        }
    };

    stage(0, 0);
    wait_async_lds();
    __syncthreads();

    int cur = 0;
    for (int k0 = 0; k0 < Kd; k0 += 32) {
        const int nxt = cur ^ 1;
        if (k0 + 32 < Kd) stage(k0 + 32, nxt);   // overlap copy of next slice

        // ---- A fragments: ISA 16-bit A 16x32 layout ----
        bf16x16 afrag[4];
        #pragma unroll
        for (int mi = 0; mi < 4; ++mi) {
            const __bf16* ap = &As[cur][(wm + mi * 16 + l16) * LDSTR + a_koff];
            bf16x8 lo = *(const bf16x8*)(ap);        // K = koff .. koff+7
            bf16x8 hi = *(const bf16x8*)(ap + 16);   // K = koff+16 .. koff+23
            afrag[mi] = __builtin_shufflevector(lo, hi, 0,1,2,3,4,5,6,7,8,9,10,11,12,13,14,15);
        }
        // ---- per-ni B fragment, then 4 WMMAs down the M tiles ----
        #pragma unroll
        for (int ni = 0; ni < 4; ++ni) {
            const __bf16* bp = &Bs[cur][(wn + ni * 16 + l16) * LDSTR + b_koff];
            bf16x8 lo = *(const bf16x8*)(bp);        // K = b_koff .. +7
            bf16x8 hi = *(const bf16x8*)(bp + 16);   // K = b_koff+16 .. +23
            bf16x16 bfrag = __builtin_shufflevector(lo, hi, 0,1,2,3,4,5,6,7,8,9,10,11,12,13,14,15);
            #pragma unroll
            for (int mi = 0; mi < 4; ++mi) {
                acc[mi][ni] = __builtin_amdgcn_wmma_f32_16x16x32_bf16(
                    false, afrag[mi], false, bfrag,
                    (short)0, acc[mi][ni], false, false);
            }
        }

        wait_async_lds();   // own async copies done before signaling
        __syncthreads();    // everyone's copies visible; reads of `cur` done
        cur = nxt;
    }

    // ---- store: D layout (VGPR r -> row r + 8*half, col = lane&15) ----
    __bf16* Cb = (__bf16*)Cg + (size_t)bz * sC;
    float*  Cf = (float*)Cg  + (size_t)bz * sC;
    #pragma unroll
    for (int mi = 0; mi < 4; ++mi) {
        #pragma unroll
        for (int ni = 0; ni < 4; ++ni) {
            const size_t col   = (size_t)(n0 + wn + ni * 16 + l16);
            const int    rbase = m0 + wm + mi * 16 + half * 8;
            #pragma unroll
            for (int r = 0; r < 8; ++r) {
                float v = acc[mi][ni][r] * scale;
                if (OUT_BF16) Cb[(size_t)(rbase + r) * Nd + col] = (__bf16)v;
                else          Cf[(size_t)(rbase + r) * Nd + col] = v;
            }
        }
    }
}

// ---------------------------------------------------------------------------
// Column softmax (axis=1 of w[b,i,j] == over query index i), causal: keep i>=j.
// One thread per (b, j): fully coalesced across j at every i-iteration.
// Writes bf16 P[b,i,j] (zeros where i<j).
// ---------------------------------------------------------------------------
__global__ __launch_bounds__(256)
void col_softmax(const float* __restrict__ S, __bf16* __restrict__ P) {
    const int j = blockIdx.x * blockDim.x + threadIdx.x;   // 0..T-1
    const int b = blockIdx.y;
    const float* s = S + (size_t)b * TDIM * TDIM;
    __bf16*      p = P + (size_t)b * TDIM * TDIM;

    float m = -3.4e38f;
    for (int i = j; i < TDIM; ++i) m = fmaxf(m, s[(size_t)i * TDIM + j]);
    float sum = 0.0f;
    for (int i = j; i < TDIM; ++i) sum += __expf(s[(size_t)i * TDIM + j] - m);
    const float inv = 1.0f / sum;
    for (int i = 0; i < TDIM; ++i) {
        float v = (i < j) ? 0.0f : __expf(s[(size_t)i * TDIM + j] - m) * inv;
        p[(size_t)i * TDIM + j] = (__bf16)v;
    }
}

// ---------------------------------------------------------------------------
extern "C" void kernel_launch(void* const* d_in, const int* in_sizes, int n_in,
                              void* d_out, int out_size, void* d_ws, size_t ws_size,
                              hipStream_t stream) {
    (void)in_sizes; (void)n_in; (void)out_size; (void)ws_size;
    const float* X  = (const float*)d_in[0];   // [B,T,E]
    const float* Wq = (const float*)d_in[1];   // [E,H]
    const float* Wk = (const float*)d_in[2];
    const float* Wv = (const float*)d_in[3];
    float* out = (float*)d_out;                // [B,T,H] fp32

    const size_t MT = (size_t)BDIM * TDIM;     // 8192 rows
    const size_t XE = MT * EDIM;               // 8,388,608
    const size_t WE = (size_t)EDIM * HDIM;     // 1,048,576
    const size_t TT = (size_t)TDIM * TDIM;     // 4,194,304
    const size_t TH = (size_t)TDIM * HDIM;     // 2,097,152

    // workspace layout (bf16 unless noted)
    __bf16* Xb  = (__bf16*)d_ws;
    __bf16* Wqb = Xb  + XE;
    __bf16* Wkb = Wqb + WE;
    __bf16* Wvb = Wkb + WE;
    __bf16* Qb  = Wvb + WE;
    __bf16* Kb  = Qb  + XE;
    __bf16* Vb  = Kb  + XE;
    float*  S   = (float*)(Vb + XE);           // [B,T,T] fp32 scores
    __bf16* P   = (__bf16*)(S + (size_t)BDIM * TT);

    // (1) convert inputs/weights to bf16
    cvt_f32_bf16<<<dim3(8192), dim3(256), 0, stream>>>(X,  Xb,  (int)XE);
    cvt_f32_bf16<<<dim3(1024), dim3(256), 0, stream>>>(Wq, Wqb, (int)WE);
    cvt_f32_bf16<<<dim3(1024), dim3(256), 0, stream>>>(Wk, Wkb, (int)WE);
    cvt_f32_bf16<<<dim3(1024), dim3(256), 0, stream>>>(Wv, Wvb, (int)WE);

    // (2) Q/K/V = Xb @ Wb   : M=8192, N=1024, K=1024
    dim3 gQKV(HDIM / 128, (unsigned)(MT / 128), 1);
    gemm_wmma_bf16<false, true><<<gQKV, 128, 0, stream>>>(Xb, Wqb, Qb, (int)MT, HDIM, EDIM, 0, 0, 0, 1.0f);
    gemm_wmma_bf16<false, true><<<gQKV, 128, 0, stream>>>(Xb, Wkb, Kb, (int)MT, HDIM, EDIM, 0, 0, 0, 1.0f);
    gemm_wmma_bf16<false, true><<<gQKV, 128, 0, stream>>>(Xb, Wvb, Vb, (int)MT, HDIM, EDIM, 0, 0, 0, 1.0f);

    // (3) S[b] = Q[b] @ K[b]^T * E^-0.5 : M=N=2048, K=1024, batched over B
    dim3 gS(TDIM / 128, TDIM / 128, BDIM);
    gemm_wmma_bf16<true, false><<<gS, 128, 0, stream>>>(Qb, Kb, S, TDIM, TDIM, HDIM,
                                                        (long long)TH, (long long)TH, (long long)TT,
                                                        0.03125f /* 1/sqrt(1024) */);

    // (4) causal mask + softmax over axis=1 (columns), write bf16 P
    col_softmax<<<dim3(TDIM / 256, BDIM), 256, 0, stream>>>(S, P);

    // (5) out[b] = P[b] @ V[b] : M=2048, N=1024, K=2048, batched over B
    dim3 gO(HDIM / 128, TDIM / 128, BDIM);
    gemm_wmma_bf16<false, false><<<gO, 128, 0, stream>>>(P, Vb, out, TDIM, HDIM, TDIM,
                                                         (long long)TT, (long long)TH, (long long)TH,
                                                         1.0f);
}
